// SparseAttention_40948218200643
// MI455X (gfx1250) — matrix operand order repaired and checked by reading, not compiled, not run on hardware
//
#include <hip/hip_runtime.h>

// ---------------------------------------------------------------------------
// Types for CDNA5 WMMA (wave32): bf16 A/B fragments (16 x bf16 = 8 VGPRs),
// fp32 C/D accumulator (8 x f32).
// ---------------------------------------------------------------------------
typedef __bf16 bf16_t;
typedef __attribute__((ext_vector_type(16))) bf16_t v16bf;
typedef __attribute__((ext_vector_type(8)))  float  v8f;

union FragAB {
    v16bf v;
    uint4 q[2];             // two 16-byte halves (elements 0..7, 8..15)
    unsigned short u[16];   // raw bf16 bits
};

__device__ __forceinline__ unsigned short f32_bf16(float f) {
    unsigned int u = __float_as_uint(f);
    u += 0x7FFFu + ((u >> 16) & 1u);   // round-to-nearest-even
    return (unsigned short)(u >> 16);
}

__device__ __forceinline__ unsigned lds_byte_off(const void* p) {
    // For LDS-space objects, flat address low 32 bits == LDS byte offset
    // (ISA 10.2 aperture rules).
    return (unsigned)(unsigned long long)(uintptr_t)p;
}

// 16-lane butterfly reductions via ds_swizzle (group-of-32 xor patterns;
// offset = xor_mask<<10 | and_mask 0x1F). Masks 1/2/4/8 stay inside each
// 16-lane half, matching the WMMA C-matrix row placement.
__device__ __forceinline__ float row16_max(float x) {
    x = fmaxf(x, __int_as_float(__builtin_amdgcn_ds_swizzle(__float_as_int(x), 0x041F)));
    x = fmaxf(x, __int_as_float(__builtin_amdgcn_ds_swizzle(__float_as_int(x), 0x081F)));
    x = fmaxf(x, __int_as_float(__builtin_amdgcn_ds_swizzle(__float_as_int(x), 0x101F)));
    x = fmaxf(x, __int_as_float(__builtin_amdgcn_ds_swizzle(__float_as_int(x), 0x201F)));
    return x;
}
__device__ __forceinline__ float row16_sum(float x) {
    x += __int_as_float(__builtin_amdgcn_ds_swizzle(__float_as_int(x), 0x041F));
    x += __int_as_float(__builtin_amdgcn_ds_swizzle(__float_as_int(x), 0x081F));
    x += __int_as_float(__builtin_amdgcn_ds_swizzle(__float_as_int(x), 0x101F));
    x += __int_as_float(__builtin_amdgcn_ds_swizzle(__float_as_int(x), 0x201F));
    return x;
}

// ---------------------------------------------------------------------------
// fp32 -> bf16 conversion
// ---------------------------------------------------------------------------
__global__ void cvt_bf16_kernel(const float* __restrict__ src,
                                unsigned short* __restrict__ dst, int n) {
    int i = blockIdx.x * blockDim.x + threadIdx.x;
    if (i < n) dst[i] = f32_bf16(src[i]);
}

// ---------------------------------------------------------------------------
// Tiled GEMM: out[M,N] = A[M,K](bf16) @ W[K,N](bf16) + bias[N]
// Block = 256 threads (8 waves), 128x128 tile, K-step 32.
// Wave grid 4(M) x 2(N): each wave owns 32x64 (2x4 grid of 16x16 WMMA tiles).
// A tile is staged memory->LDS with GLOBAL_LOAD_ASYNC_TO_LDS_B128 (ASYNCcnt).
// ---------------------------------------------------------------------------
template <bool OUT_F32>
__global__ __launch_bounds__(256) void gemm_bias_kernel(
    const unsigned short* __restrict__ A,   // [Mtot,K] bf16 bits
    const unsigned short* __restrict__ W,   // [K,N]    bf16 bits
    const float* __restrict__ bias,         // [N]
    void* __restrict__ out, int Mtot, int N, int K)
{
    __shared__ __align__(16) unsigned short Alds[128 * 32];   // [row][k]
    __shared__ __align__(16) unsigned short BT[128 * 32];     // [n][k] (transposed W)

    const int tid  = threadIdx.x;
    const int lane = tid & 31;
    const int wave = tid >> 5;
    const int wm   = wave & 3;      // 0..3 -> M offset wm*32
    const int wn   = wave >> 2;     // 0..1 -> N offset wn*64
    const int lh   = lane >> 4;     // lane half (0/1)
    const int ll   = lane & 15;

    const int m0 = blockIdx.y * 128;
    const int n0 = blockIdx.x * 128;

    v8f acc[2][4];
#pragma unroll
    for (int i = 0; i < 2; i++)
#pragma unroll
        for (int j = 0; j < 4; j++)
#pragma unroll
            for (int r = 0; r < 8; r++) acc[i][j][r] = 0.f;

    for (int kb = 0; kb < K; kb += 32) {
        // ---- stage A tile: async memory -> LDS, 16B per lane per issue ----
#pragma unroll
        for (int i = 0; i < 2; i++) {
            int c   = tid + i * 256;          // 512 chunks
            int row = c >> 2;
            int k8  = (c & 3) * 8;
            unsigned goff = (unsigned)(((m0 + row) * K + kb + k8) * 2);
            unsigned loff = lds_byte_off(&Alds[row * 32 + k8]);
            asm volatile("global_load_async_to_lds_b128 %0, %1, %2"
                         :: "v"(loff), "v"(goff), "s"(A) : "memory");
        }
        // ---- stage W tile transposed: BT[n][k] ----
#pragma unroll
        for (int i = 0; i < 2; i++) {
            int c  = tid + i * 256;
            int kr = c >> 4;                  // 0..31
            int n8 = (c & 15) * 8;            // 0..120
            uint4 d = *reinterpret_cast<const uint4*>(
                W + (size_t)(kb + kr) * N + n0 + n8);
            const unsigned short* e = reinterpret_cast<const unsigned short*>(&d);
#pragma unroll
            for (int j = 0; j < 8; j++)
                BT[(n8 + j) * 32 + kr] = e[j];
        }
        asm volatile("s_wait_asynccnt 0" ::: "memory");
        __syncthreads();

        // ---- fragments + WMMA ----
        FragAB fa[2], fb[4];
#pragma unroll
        for (int mt = 0; mt < 2; mt++) {
            int row = wm * 32 + mt * 16 + ll;
            fa[mt].q[0] = *reinterpret_cast<const uint4*>(&Alds[row * 32 + lh * 8]);
            fa[mt].q[1] = *reinterpret_cast<const uint4*>(&Alds[row * 32 + 16 + lh * 8]);
        }
#pragma unroll
        for (int nt = 0; nt < 4; nt++) {
            int n = wn * 64 + nt * 16 + ll;
            fb[nt].q[0] = *reinterpret_cast<const uint4*>(&BT[n * 32 + lh * 8]);
            fb[nt].q[1] = *reinterpret_cast<const uint4*>(&BT[n * 32 + 16 + lh * 8]);
        }
#pragma unroll
        for (int mt = 0; mt < 2; mt++)
#pragma unroll
            for (int nt = 0; nt < 4; nt++)
                acc[mt][nt] = __builtin_amdgcn_wmma_f32_16x16x32_bf16(
                    false, fa[mt].v, false, fb[nt].v,
                    (short)0, acc[mt][nt], false, false);
        __syncthreads();
    }

    // ---- epilogue: C layout -> lane holds col ll, VGPR r -> row lh*8+r ----
#pragma unroll
    for (int mt = 0; mt < 2; mt++) {
#pragma unroll
        for (int nt = 0; nt < 4; nt++) {
            int n   = n0 + wn * 64 + nt * 16 + ll;
            float b = bias[n];
#pragma unroll
            for (int r = 0; r < 8; r++) {
                int m   = m0 + wm * 32 + mt * 16 + lh * 8 + r;
                float v = acc[mt][nt][r] + b;
                if (OUT_F32)
                    reinterpret_cast<float*>(out)[(size_t)m * N + n] = v;
                else
                    reinterpret_cast<unsigned short*>(out)[(size_t)m * N + n] =
                        f32_bf16(v);
            }
        }
    }
}

// ---------------------------------------------------------------------------
// One 32-key S-tile step of flash attention.
// MASKED=false: memory-prefix phase (s < 512): K == V == memory rows,
//               single global load feeds both tiles, no mask math.
// MASKED=true : causal tail (s >= 512): K/V from qkv, per-element mask.
// Softmax scale is pre-folded into the Q fragment (exact pow2).
// ---------------------------------------------------------------------------
template <bool MASKED>
static __device__ __forceinline__ void attn_tile(
    int st, int b, int h, int q0, int tid, int wave, int lh, int ll,
    const unsigned short* __restrict__ qkvb,
    const unsigned short* __restrict__ memb,
    unsigned short* Klds, unsigned short* VT, unsigned short* Plds,
    const FragAB (&qf)[2], v8f (&ao)[4], float (&mrow)[8], float (&lrow)[8])
{
    // ---- cooperative stage of K tile [32][64] and V^T tile [64][32] ----
#pragma unroll
    for (int i = 0; i < 2; i++) {
        int c   = tid + i * 128;          // 256 chunks of 8 elems
        int key = c >> 3;
        int d8  = (c & 7) * 8;
        int s   = st * 32 + key;
        if (MASKED) {
            size_t rb = ((size_t)(b * 256 + s - 512)) * 1152 + h * 64 + d8;
            uint4 kv = *reinterpret_cast<const uint4*>(qkvb + rb + 384);
            *reinterpret_cast<uint4*>(&Klds[key * 64 + d8]) = kv;
            uint4 vv = *reinterpret_cast<const uint4*>(qkvb + rb + 768);
            const unsigned short* ve = reinterpret_cast<const unsigned short*>(&vv);
#pragma unroll
            for (int j = 0; j < 8; j++) VT[(d8 + j) * 32 + key] = ve[j];
        } else {
            // memory prefix: K and V are the same rows
            uint4 kv = *reinterpret_cast<const uint4*>(
                memb + (size_t)s * 384 + h * 64 + d8);
            *reinterpret_cast<uint4*>(&Klds[key * 64 + d8]) = kv;
            const unsigned short* ve = reinterpret_cast<const unsigned short*>(&kv);
#pragma unroll
            for (int j = 0; j < 8; j++) VT[(d8 + j) * 32 + key] = ve[j];
        }
    }
    __syncthreads();

    // ---- scores: S = (Q*scale) K^T, two 16-key subtiles ----
    v8f sc[2];
#pragma unroll
    for (int nt = 0; nt < 2; nt++) {
        v8f a;
#pragma unroll
        for (int r = 0; r < 8; r++) a[r] = 0.f;
#pragma unroll
        for (int db = 0; db < 2; db++) {
            FragAB kf;
            kf.q[0] = *reinterpret_cast<const uint4*>(
                &Klds[(nt * 16 + ll) * 64 + db * 32 + lh * 8]);
            kf.q[1] = *reinterpret_cast<const uint4*>(
                &Klds[(nt * 16 + ll) * 64 + db * 32 + 16 + lh * 8]);
            a = __builtin_amdgcn_wmma_f32_16x16x32_bf16(
                false, qf[db].v, false, kf.v, (short)0, a, false, false);
        }
        if (MASKED) {
            int srel = st * 32 + nt * 16 + ll - 512;   // key index in causal tail
#pragma unroll
            for (int r = 0; r < 8; r++) {
                int qi = q0 + lh * 8 + r;
                a[r] = (srel <= qi) ? a[r] : -1e30f;
            }
        }
        sc[nt] = a;
    }

    // ---- online softmax: batch all 8 row maxes first (ILP), then exp/sum ----
    float tmax[8];
#pragma unroll
    for (int r = 0; r < 8; r++)
        tmax[r] = row16_max(fmaxf(sc[0][r], sc[1][r]));

    float corr[8];
#pragma unroll
    for (int r = 0; r < 8; r++) {
        float mn = fmaxf(mrow[r], tmax[r]);
        corr[r]  = __expf(mrow[r] - mn);
        float p0 = __expf(sc[0][r] - mn);
        float p1 = __expf(sc[1][r] - mn);
        mrow[r]  = mn;
        int m = lh * 8 + r;                 // C layout row
        Plds[wave * 512 + m * 32 + ll]      = f32_bf16(p0);
        Plds[wave * 512 + m * 32 + 16 + ll] = f32_bf16(p1);
        float s  = row16_sum(p0 + p1);
        lrow[r]  = lrow[r] * corr[r] + s;
    }
#pragma unroll
    for (int dt = 0; dt < 4; dt++)
#pragma unroll
        for (int r = 0; r < 8; r++) ao[dt][r] *= corr[r];
    __syncthreads();

    // ---- O += P @ V (A frag re-read from LDS in A layout) ----
    FragAB pf;
    pf.q[0] = *reinterpret_cast<const uint4*>(&Plds[wave * 512 + ll * 32 + lh * 8]);
    pf.q[1] = *reinterpret_cast<const uint4*>(&Plds[wave * 512 + ll * 32 + 16 + lh * 8]);
#pragma unroll
    for (int dt = 0; dt < 4; dt++) {
        FragAB vf;
        vf.q[0] = *reinterpret_cast<const uint4*>(
            &VT[(dt * 16 + ll) * 32 + lh * 8]);
        vf.q[1] = *reinterpret_cast<const uint4*>(
            &VT[(dt * 16 + ll) * 32 + 16 + lh * 8]);
        ao[dt] = __builtin_amdgcn_wmma_f32_16x16x32_bf16(
            false, pf.v, false, vf.v, (short)0, ao[dt], false, false);
    }
    __syncthreads();
}

// ---------------------------------------------------------------------------
// Flash attention with memory prefix.
//   qkv   : [B*T, 1152] bf16  (q:0..383, k:384..767, v:768..1151)
//   memb  : [512, 384]  bf16  (shared K=V prefix)
//   yb    : [B*T, 384]  bf16  attention output
// Grid: B*H*(T/64) blocks of 128 threads (4 waves; wave = 16 q rows).
// Phase 1: 16 unmasked prefix tiles. Phase 2: up to 2*qb+2 causal tiles
// (uniform per block -> barrier-safe; fully-masked tiles skipped).
// ---------------------------------------------------------------------------
__global__ __launch_bounds__(128) void attn_kernel(
    const unsigned short* __restrict__ qkvb,
    const unsigned short* __restrict__ memb,
    unsigned short* __restrict__ yb)
{
    __shared__ __align__(16) unsigned short Klds[32 * 64];      // [key][d]
    __shared__ __align__(16) unsigned short VT[64 * 32];        // [d][key]
    __shared__ __align__(16) unsigned short Plds[4 * 16 * 32];  // per-wave P tile

    const int tid  = threadIdx.x;
    const int lane = tid & 31;
    const int wave = tid >> 5;
    const int lh   = lane >> 4;
    const int ll   = lane & 15;

    int bx = blockIdx.x;
    const int qb = bx & 3;  bx >>= 2;
    const int h  = bx % 6;
    const int b  = bx / 6;
    const int q0 = qb * 64 + wave * 16;       // wave's first q row (0..255)

    // ---- Q A-fragments (d = 0..31, 32..63), scale folded in (exact /8) ----
    FragAB qf[2];
    {
        const size_t rowbase = ((size_t)(b * 256 + q0 + ll)) * 1152 + h * 64;
#pragma unroll
        for (int f = 0; f < 2; f++) {
            qf[f].q[0] = *reinterpret_cast<const uint4*>(qkvb + rowbase + f * 32 + lh * 8);
            qf[f].q[1] = *reinterpret_cast<const uint4*>(qkvb + rowbase + f * 32 + 16 + lh * 8);
#pragma unroll
            for (int e = 0; e < 16; e++) {
                float v = __uint_as_float((unsigned)qf[f].u[e] << 16) * 0.125f;
                qf[f].u[e] = f32_bf16(v);      // exact: pow2 scale
            }
        }
    }

    float mrow[8], lrow[8];
    v8f   ao[4];
#pragma unroll
    for (int r = 0; r < 8; r++) { mrow[r] = -1e30f; lrow[r] = 0.f; }
#pragma unroll
    for (int dt = 0; dt < 4; dt++)
#pragma unroll
        for (int r = 0; r < 8; r++) ao[dt][r] = 0.f;

    // Phase 1: memory prefix (keys 0..511), fully visible.
    for (int st = 0; st < 16; ++st)
        attn_tile<false>(st, b, h, q0, tid, wave, lh, ll, qkvb, memb,
                         Klds, VT, Plds, qf, ao, mrow, lrow);
    // Phase 2: causal tail (keys 512..). Tile j needs 32*j <= qb*64 + 63.
    const int st_stop = 16 + 2 * qb + 2;
    for (int st = 16; st < st_stop; ++st)
        attn_tile<true>(st, b, h, q0, tid, wave, lh, ll, qkvb, memb,
                        Klds, VT, Plds, qf, ao, mrow, lrow);

    // ---- normalize + store y (bf16) ----
#pragma unroll
    for (int dt = 0; dt < 4; dt++) {
#pragma unroll
        for (int r = 0; r < 8; r++) {
            int m  = lh * 8 + r;
            int qi = q0 + m;
            yb[((size_t)(b * 256 + qi)) * 384 + h * 64 + dt * 16 + ll] =
                f32_bf16(ao[dt][r] / lrow[r]);
        }
    }
}

// ---------------------------------------------------------------------------
// Host launcher
// Inputs: x[64,256,384], memory[512,384], w_qkv[384,1152], b_qkv[1152],
//         w_proj[384,384], b_proj[384]  (all fp32). Output fp32 [64,256,384].
// ---------------------------------------------------------------------------
extern "C" void kernel_launch(void* const* d_in, const int* in_sizes, int n_in,
                              void* d_out, int out_size, void* d_ws, size_t ws_size,
                              hipStream_t stream) {
    const float* x      = (const float*)d_in[0];
    const float* memory = (const float*)d_in[1];
    const float* w_qkv  = (const float*)d_in[2];
    const float* b_qkv  = (const float*)d_in[3];
    const float* w_proj = (const float*)d_in[4];
    const float* b_proj = (const float*)d_in[5];

    char* ws = (char*)d_ws;
    size_t off = 0;
    unsigned short* xb   = (unsigned short*)(ws + off); off += (size_t)16384 * 384 * 2;
    unsigned short* wqb  = (unsigned short*)(ws + off); off += (size_t)384 * 1152 * 2;
    unsigned short* wpb  = (unsigned short*)(ws + off); off += (size_t)384 * 384 * 2;
    unsigned short* mb   = (unsigned short*)(ws + off); off += (size_t)512 * 384 * 2;
    unsigned short* qkvb = (unsigned short*)(ws + off); off += (size_t)16384 * 1152 * 2;
    unsigned short* yb   = (unsigned short*)(ws + off);

    // fp32 -> bf16 conversions
    {
        int n;
        n = 16384 * 384;
        cvt_bf16_kernel<<<(n + 255) / 256, 256, 0, stream>>>(x, xb, n);
        n = 384 * 1152;
        cvt_bf16_kernel<<<(n + 255) / 256, 256, 0, stream>>>(w_qkv, wqb, n);
        n = 384 * 384;
        cvt_bf16_kernel<<<(n + 255) / 256, 256, 0, stream>>>(w_proj, wpb, n);
        n = 512 * 384;
        cvt_bf16_kernel<<<(n + 255) / 256, 256, 0, stream>>>(memory, mb, n);
    }

    // qkv = x @ w_qkv + b_qkv  (bf16 out)
    gemm_bias_kernel<false><<<dim3(1152 / 128, 16384 / 128), 256, 0, stream>>>(
        xb, wqb, b_qkv, (void*)qkvb, 16384, 1152, 384);

    // flash attention with 512-row memory prefix + causal tail
    attn_kernel<<<dim3(64 * 6 * 4), 128, 0, stream>>>(qkvb, mb, yb);

    // out = y @ w_proj + b_proj  (fp32 out)
    gemm_bias_kernel<true><<<dim3(384 / 128, 16384 / 128), 256, 0, stream>>>(
        yb, wpb, b_proj, d_out, 16384, 384, 384);
}
